// pmfDiT_65893388255874
// MI455X (gfx1250) — matrix operand order repaired and checked by reading, not compiled, not run on hardware
//
#include <hip/hip_runtime.h>
#include <hip/hip_bf16.h>

typedef __attribute__((ext_vector_type(16))) _Float16 v16h;
typedef __attribute__((ext_vector_type(8)))  _Float16 v8h;
typedef __attribute__((ext_vector_type(8)))  float    v8f;
typedef unsigned int u32x4 __attribute__((ext_vector_type(4)));
typedef int          s32x8 __attribute__((ext_vector_type(8)));
typedef int          s32x4 __attribute__((ext_vector_type(4)));

constexpr int cB = 8, cS = 1042, cH = 768, cNH = 12, cHD = 64, cP = 1024, cMLP = 2048;
constexpr int cNT = cB * cS;            // 8336 tokens
constexpr float cEPS = 1e-6f;

// ---------------------------------------------------------------------------
// Tensor Data Mover: 2D f16 tile (tile_d0 contiguous halves x tile_d1 rows)
// global -> LDS, row stride stride_d0 (halves).  OOB rows/cols zero-fill.
// D# packing per CDNA5 ISA 8.3/8.4: group0 = {count|flags, lds_addr,
// global_addr[31:0], global_addr[56:32]|type=2}; group1 holds data_size,
// tensor/tile dims and dim0 stride.  Issue from ONE wave per workgroup.
// ---------------------------------------------------------------------------
#if defined(__gfx1250__) && __has_builtin(__builtin_amdgcn_tensor_load_to_lds)
#define HAVE_TDM 1
#endif

__device__ __forceinline__ void tdm_load_2d(unsigned lds_addr, const void* gptr,
                                            unsigned tile_d0, unsigned tile_d1,
                                            unsigned tensor_d0, unsigned tensor_d1,
                                            unsigned stride_d0) {
#if defined(HAVE_TDM)
  unsigned long long ga = (unsigned long long)(uintptr_t)gptr;
  u32x4 g0;
  g0[0] = 1u;                                             // count=1, user mode
  g0[1] = lds_addr;                                       // LDS byte address
  g0[2] = (unsigned)(ga & 0xFFFFFFFFu);                   // global_addr lo
  g0[3] = (unsigned)((ga >> 32) & 0x1FFFFFFu) | (2u << 30); // addr hi | type=2
  s32x8 g1;
  g1[0] = (int)(1u << 16);                                // data_size=1 (2B)
  g1[1] = (int)((tensor_d0 & 0xFFFFu) << 16);             // tensor_dim0[15:0]
  g1[2] = (int)(((tensor_d0 >> 16) & 0xFFFFu) |
                ((tensor_d1 & 0xFFFFu) << 16));           // dim0 hi | dim1 lo
  g1[3] = (int)(((tensor_d1 >> 16) & 0xFFFFu) |
                ((tile_d0 & 0xFFFFu) << 16));             // dim1 hi | tile0
  g1[4] = (int)(tile_d1 & 0xFFFFu);                       // tile1 (tile2=0)
  g1[5] = (int)stride_d0;                                 // dim0 stride lo32
  g1[6] = 0;                                              // stride hi / dim1 stride
  g1[7] = 0;
  s32x4 z4 = {0, 0, 0, 0};
#if __clang_major__ >= 23
  s32x8 z8 = {0, 0, 0, 0, 0, 0, 0, 0};
  __builtin_amdgcn_tensor_load_to_lds(g0, g1, z4, z4, z8, 0);
#else
  __builtin_amdgcn_tensor_load_to_lds(g0, g1, z4, z4, 0);
#endif
#else
  (void)lds_addr; (void)gptr; (void)tile_d0; (void)tile_d1;
  (void)tensor_d0; (void)tensor_d1; (void)stride_d0;
#endif
}

__device__ __forceinline__ void tdm_wait() {
#if defined(HAVE_TDM)
  __builtin_amdgcn_s_wait_tensorcnt(0);
#endif
}

// ---------------------------------------------------------------------------
// f32 -> f16 weight conversion
// ---------------------------------------------------------------------------
__global__ __launch_bounds__(256) void f32_to_f16_kernel(const float* __restrict__ in,
                                                         _Float16* __restrict__ out, int n) {
  int i = blockIdx.x * 256 + threadIdx.x;
  if (i < n) out[i] = (_Float16)in[i];
}

// ---------------------------------------------------------------------------
// RMSNorm over H=768, one block (256 thr) per token, writes f16
// ---------------------------------------------------------------------------
__global__ __launch_bounds__(256) void rmsnorm_kernel(const float* __restrict__ x,
                                                      const float* __restrict__ w,
                                                      _Float16* __restrict__ out) {
  int t = blockIdx.x;
  const float* xp = x + (size_t)t * cH;
  float vals[3];
  float ss = 0.f;
#pragma unroll
  for (int i = 0; i < 3; i++) {
    vals[i] = xp[threadIdx.x + i * 256];
    ss += vals[i] * vals[i];
  }
#pragma unroll
  for (int off = 16; off; off >>= 1) ss += __shfl_xor(ss, off);
  __shared__ float red[8];
  int wave = threadIdx.x >> 5, lane = threadIdx.x & 31;
  if (lane == 0) red[wave] = ss;
  __syncthreads();
  float tot = 0.f;
#pragma unroll
  for (int i = 0; i < 8; i++) tot += red[i];
  float r = rsqrtf(tot / (float)cH + cEPS);
  _Float16* op = out + (size_t)t * cH;
#pragma unroll
  for (int i = 0; i < 3; i++) {
    int j = threadIdx.x + i * 256;
    op[j] = (_Float16)(vals[i] * r * w[j]);
  }
}

// ---------------------------------------------------------------------------
// Per-head RMSNorm + RoPE (+ optional 1/sqrt(HD) scale), in-place on f16 q/k.
// ---------------------------------------------------------------------------
__global__ __launch_bounds__(128) void qk_post_kernel(_Float16* __restrict__ qk,
                                                      const float* __restrict__ nw,
                                                      const float* __restrict__ rc,
                                                      const float* __restrict__ rs,
                                                      float outscale) {
  int wave = threadIdx.x >> 5, lane = threadIdx.x & 31;
  int g = blockIdx.x * 4 + wave;
  if (g >= cNT * cNH) return;
  int t = g / cNH, hd = g % cNH;
  int s = t % cS;
  _Float16* p = qk + (size_t)t * cH + hd * cHD;
  float x0 = (float)p[2 * lane], x1 = (float)p[2 * lane + 1];
  float ss = x0 * x0 + x1 * x1;
#pragma unroll
  for (int off = 16; off; off >>= 1) ss += __shfl_xor(ss, off);
  float r = rsqrtf(ss / (float)cHD + cEPS);
  x0 *= r * nw[2 * lane];
  x1 *= r * nw[2 * lane + 1];
  int pos = s - (cS - cP);
  if (pos >= 0) {
    float c = rc[pos * (cHD / 2) + lane], sn = rs[pos * (cHD / 2) + lane];
    float y0 = x0 * c - x1 * sn;
    float y1 = x0 * sn + x1 * c;
    x0 = y0; x1 = y1;
  }
  p[2 * lane]     = (_Float16)(x0 * outscale);
  p[2 * lane + 1] = (_Float16)(x1 * outscale);
}

// ---------------------------------------------------------------------------
// gate = sigmoid(h @ gate_W + gate_b); one wave per (token, head)
// ---------------------------------------------------------------------------
__global__ __launch_bounds__(128) void gate_kernel(const _Float16* __restrict__ h,
                                                   const float* __restrict__ gW,
                                                   const float* __restrict__ gb,
                                                   float* __restrict__ gate) {
  int wave = threadIdx.x >> 5, lane = threadIdx.x & 31;
  int g = blockIdx.x * 4 + wave;
  if (g >= cNT * cNH) return;
  int t = g / cNH, hd = g % cNH;
  const _Float16* hp = h + (size_t)t * cH;
  float acc = 0.f;
  for (int j = lane; j < cH; j += 32) acc += (float)hp[j] * gW[j * cNH + hd];
#pragma unroll
  for (int off = 16; off; off >>= 1) acc += __shfl_xor(acc, off);
  if (lane == 0) gate[g] = 1.0f / (1.0f + __expf(-(acc + gb[hd])));
}

// ---------------------------------------------------------------------------
// Generic f16 WMMA GEMM:  C[M x N] = A[M x K] @ Bw[K x N]
//   Block 256 thr (8 waves), tile 64x128; wave tile 32x32 (2x2 wmma frags).
//   A tile staged by TDM (row-major 64x32, OOB rows auto-zero); B tile
//   manually transposed into [col][k] so B-fragments are ds_load_b128s.
// EPI==0: store f16.  EPI==1: Cf = resid + acc*cscale[col]  (f32 out).
// Requires K%32==0, N%128==0; M ragged (guarded / TDM-zero-filled).
// ---------------------------------------------------------------------------
template <int EPI>
__global__ __launch_bounds__(256) void gemm_kernel(const _Float16* __restrict__ A,
                                                   const _Float16* __restrict__ Bw,
                                                   _Float16* __restrict__ Ch,
                                                   float* __restrict__ Cf,
                                                   const float* __restrict__ resid,
                                                   const float* __restrict__ cscale,
                                                   int M, int K, int N) {
  __shared__ _Float16 As[64 * 32];     // [row][k], row stride 32
  __shared__ _Float16 Bs[128 * 32];    // [col][k], col stride 32 (transposed)
  int tid = threadIdx.x;
  int lane = tid & 31, wave = tid >> 5;
  int lh = lane & 15, sel = lane >> 4;
  int bm = blockIdx.y * 64, bn = blockIdx.x * 128;
  int wm = (wave >> 2) * 32, wn = (wave & 3) * 32;

  v8f acc[2][2] = {};

  int ar = tid >> 2, ac = (tid & 3) * 8;    // A fallback staging
  int br = tid >> 3, bc = (tid & 7) * 16;   // B staging: 32 rows x 8x16 halves
  int garow = bm + ar;
  unsigned lds_as = (unsigned)(uintptr_t)(void*)As;

  for (int k0 = 0; k0 < K; k0 += 32) {
#if defined(HAVE_TDM)
    // ---- stage A tile via Tensor Data Mover (wave 0 issues the DMA) ----
    if (wave == 0) {
      tdm_load_2d(lds_as, A + (size_t)bm * K + k0,
                  /*tile*/ 32, 64,
                  /*tensor*/ (unsigned)(K - k0), (unsigned)(M - bm),
                  /*stride*/ (unsigned)K);
    }
#else
    v8h av;
    if (garow < M) {
      av = *(const v8h*)(A + (size_t)garow * K + k0 + ac);
    } else {
#pragma unroll
      for (int j = 0; j < 8; j++) av[j] = (_Float16)0.f;
    }
    *(v8h*)(As + ar * 32 + ac) = av;
#endif
    // ---- stage B tile (transpose into [col][k]) ----
    const _Float16* bp = Bw + (size_t)(k0 + br) * N + bn + bc;
    v8h b0 = *(const v8h*)bp;
    v8h b1 = *(const v8h*)(bp + 8);
#pragma unroll
    for (int j = 0; j < 8; j++) {
      Bs[(bc + j) * 32 + br]     = b0[j];
      Bs[(bc + 8 + j) * 32 + br] = b1[j];
    }
    if (k0 + 32 < K)  // prefetch next B K-tile -> global_prefetch_b8
      __builtin_prefetch(Bw + (size_t)(k0 + 32 + br) * N + bn + bc, 0, 0);
#if defined(HAVE_TDM)
    if (wave == 0) tdm_wait();
#endif
    __syncthreads();

    // ---- fragments ----
    v16h af[2], bf[2];
#pragma unroll
    for (int mi = 0; mi < 2; mi++) {
      const _Float16* ap = As + (wm + mi * 16 + lh) * 32 + sel * 8;
      v8h lo = *(const v8h*)ap, hi = *(const v8h*)(ap + 16);
#pragma unroll
      for (int j = 0; j < 8; j++) { af[mi][j] = lo[j]; af[mi][8 + j] = hi[j]; }
    }
#pragma unroll
    for (int ni = 0; ni < 2; ni++) {
      const _Float16* bpp = Bs + (wn + ni * 16 + lh) * 32 + sel * 8;
      v8h lo = *(const v8h*)bpp, hi = *(const v8h*)(bpp + 16);
#pragma unroll
      for (int j = 0; j < 8; j++) { bf[ni][j] = lo[j]; bf[ni][8 + j] = hi[j]; }
    }
#pragma unroll
    for (int mi = 0; mi < 2; mi++)
#pragma unroll
      for (int ni = 0; ni < 2; ni++)
        acc[mi][ni] = __builtin_amdgcn_wmma_f32_16x16x32_f16(
            false, af[mi], false, bf[ni], (short)0, acc[mi][ni], false, false);
    __syncthreads();
  }
  (void)garow;

  // ---- epilogue (C layout: VGPR i -> row i + 8*sel, col = lane%16) ----
#pragma unroll
  for (int mi = 0; mi < 2; mi++)
#pragma unroll
    for (int ni = 0; ni < 2; ni++)
#pragma unroll
      for (int i = 0; i < 8; i++) {
        int row = bm + wm + mi * 16 + i + sel * 8;
        int col = bn + wn + ni * 16 + lh;
        if (row < M) {
          size_t idx = (size_t)row * N + col;
          float vv = acc[mi][ni][i];
          if (EPI == 0) Ch[idx] = (_Float16)vv;
          else          Cf[idx] = resid[idx] + vv * cscale[col];
        }
      }
}

// ---------------------------------------------------------------------------
// Flash attention, wave-level: each wave owns a 16-query tile of one (b,h);
// 4 waves/block share K/V LDS tiles (32 keys/step).  K tile staged by TDM
// (keys >= S auto-zero, then masked); V manually transposed.  Scores and
// P@V both on v_wmma_f32_16x16x32_f16.
// ---------------------------------------------------------------------------
__global__ __launch_bounds__(128) void attn_kernel(const _Float16* __restrict__ q,
                                                   const _Float16* __restrict__ k,
                                                   const _Float16* __restrict__ v,
                                                   const float* __restrict__ gate,
                                                   _Float16* __restrict__ out) {
  __shared__ _Float16 Kt[32 * 64];       // [key][d]
  __shared__ _Float16 Vt[64 * 32];       // [d][key]  (transposed)
  __shared__ _Float16 Ps[4 * 16 * 32];   // wave-private P tiles

  int b = blockIdx.z, hh = blockIdx.y;
  int tid = threadIdx.x, lane = tid & 31, wave = tid >> 5;
  int lh = lane & 15, sel = lane >> 4;
  int m0 = blockIdx.x * 64 + wave * 16;
  unsigned lds_kt = (unsigned)(uintptr_t)(void*)Kt;

  // Q fragments for the whole key loop (A layout), rows clamped to S-1
  int qs = m0 + lh; if (qs > cS - 1) qs = cS - 1;
  const _Float16* qp = q + (size_t)(b * cS + qs) * cH + hh * cHD;
  v16h qa[2];
#pragma unroll
  for (int dblk = 0; dblk < 2; dblk++) {
    const _Float16* ap = qp + dblk * 32 + sel * 8;
    v8h lo = *(const v8h*)ap, hi = *(const v8h*)(ap + 16);
#pragma unroll
    for (int j = 0; j < 8; j++) { qa[dblk][j] = lo[j]; qa[dblk][8 + j] = hi[j]; }
  }

  v8f o[4] = {};
  float mrow[8], lrow[8];
#pragma unroll
  for (int i = 0; i < 8; i++) { mrow[i] = -1e30f; lrow[i] = 0.f; }

  int stgKey = tid >> 2;          // 0..31
  int stgC   = (tid & 3) * 16;    // 0,16,32,48

  for (int kb = 0; kb < cS; kb += 32) {
    // ---- stage K tile via TDM; V transposed manually ----
#if defined(HAVE_TDM)
    if (wave == 0) {
      tdm_load_2d(lds_kt, k + (size_t)(b * cS + kb) * cH + hh * cHD,
                  /*tile*/ 64, 32,
                  /*tensor*/ 64, (unsigned)(cS - kb),
                  /*stride*/ (unsigned)cH);
    }
#endif
    int ks = kb + stgKey;
    int ksc = ks < cS ? ks : cS - 1;
#if !defined(HAVE_TDM)
    const _Float16* kp = k + (size_t)(b * cS + ksc) * cH + hh * cHD + stgC;
    v8h k0v = *(const v8h*)kp, k1v = *(const v8h*)(kp + 8);
    *(v8h*)(Kt + stgKey * 64 + stgC)     = k0v;
    *(v8h*)(Kt + stgKey * 64 + stgC + 8) = k1v;
#endif
    const _Float16* vp = v + (size_t)(b * cS + ksc) * cH + hh * cHD + stgC;
    v8h v0v = *(const v8h*)vp, v1v = *(const v8h*)(vp + 8);
#pragma unroll
    for (int j = 0; j < 8; j++) {
      Vt[(stgC + j) * 32 + stgKey]     = v0v[j];
      Vt[(stgC + 8 + j) * 32 + stgKey] = v1v[j];
    }
#if defined(HAVE_TDM)
    if (wave == 0) tdm_wait();
#endif
    __syncthreads();

    // ---- scores = Q @ K^T  (two 16-key subtiles, K-dim = HD = 2x32) ----
    v8f sc[2];
#pragma unroll
    for (int ksub = 0; ksub < 2; ksub++) {
      v8f c = {};
#pragma unroll
      for (int dblk = 0; dblk < 2; dblk++) {
        const _Float16* bp = Kt + (ksub * 16 + lh) * 64 + dblk * 32 + sel * 8;
        v8h lo = *(const v8h*)bp, hi = *(const v8h*)(bp + 16);
        v16h bfr;
#pragma unroll
        for (int j = 0; j < 8; j++) { bfr[j] = lo[j]; bfr[8 + j] = hi[j]; }
        c = __builtin_amdgcn_wmma_f32_16x16x32_f16(false, qa[dblk], false, bfr,
                                                   (short)0, c, false, false);
      }
      int keyg = kb + ksub * 16 + lh;
      if (keyg >= cS) {
#pragma unroll
        for (int i = 0; i < 8; i++) c[i] = -1e30f;
      }
      sc[ksub] = c;
    }

    // ---- online softmax (row = i + 8*sel; reduce across 16-lane group) ----
    float pv[2][8];
#pragma unroll
    for (int i = 0; i < 8; i++) {
      float t = fmaxf(sc[0][i], sc[1][i]);
#pragma unroll
      for (int off = 1; off < 16; off <<= 1) t = fmaxf(t, __shfl_xor(t, off));
      float mnew = fmaxf(mrow[i], t);
      float alpha = __expf(mrow[i] - mnew);
      float p0 = __expf(sc[0][i] - mnew);
      float p1 = __expf(sc[1][i] - mnew);
      float rs = p0 + p1;
#pragma unroll
      for (int off = 1; off < 16; off <<= 1) rs += __shfl_xor(rs, off);
      lrow[i] = alpha * lrow[i] + rs;
      mrow[i] = mnew;
      pv[0][i] = p0; pv[1][i] = p1;
#pragma unroll
      for (int d = 0; d < 4; d++) o[d][i] *= alpha;
    }

    // ---- P: C layout -> A layout via wave-private LDS tile ----
    _Float16* Pw = Ps + wave * 512;
#pragma unroll
    for (int ksub = 0; ksub < 2; ksub++)
#pragma unroll
      for (int i = 0; i < 8; i++)
        Pw[(i + 8 * sel) * 32 + ksub * 16 + lh] = (_Float16)pv[ksub][i];

    v16h pf;
    {
      const _Float16* pp = Pw + lh * 32 + sel * 8;
      v8h lo = *(const v8h*)pp, hi = *(const v8h*)(pp + 16);
#pragma unroll
      for (int j = 0; j < 8; j++) { pf[j] = lo[j]; pf[8 + j] = hi[j]; }
    }

    // ---- O += P @ V  (4 d-chunks of 16, K-dim = 32 keys) ----
#pragma unroll
    for (int d = 0; d < 4; d++) {
      const _Float16* bp = Vt + (d * 16 + lh) * 32 + sel * 8;
      v8h lo = *(const v8h*)bp, hi = *(const v8h*)(bp + 16);
      v16h bfr;
#pragma unroll
      for (int j = 0; j < 8; j++) { bfr[j] = lo[j]; bfr[8 + j] = hi[j]; }
      o[d] = __builtin_amdgcn_wmma_f32_16x16x32_f16(false, pf, false, bfr,
                                                    (short)0, o[d], false, false);
    }
    __syncthreads();
  }

  // ---- epilogue: normalize, gate, store f16 ----
#pragma unroll
  for (int i = 0; i < 8; i++) {
    int srow = m0 + i + 8 * sel;
    if (srow < cS) {
      float g = gate[(size_t)(b * cS + srow) * cNH + hh];
      float inv = 1.0f / lrow[i];
#pragma unroll
      for (int d = 0; d < 4; d++)
        out[(size_t)(b * cS + srow) * cH + hh * cHD + d * 16 + lh] =
            (_Float16)(o[d][i] * inv * g);
    }
  }
}

// ---------------------------------------------------------------------------
// SwiGLU elementwise: mlp = silu(g) * u  (f16 in, f16 out)
// ---------------------------------------------------------------------------
__global__ __launch_bounds__(256) void swiglu_kernel(const _Float16* __restrict__ g,
                                                     const _Float16* __restrict__ u,
                                                     _Float16* __restrict__ out, int n) {
  int i = blockIdx.x * 256 + threadIdx.x;
  if (i < n) {
    float gv = (float)g[i], uv = (float)u[i];
    out[i] = (_Float16)(gv / (1.f + __expf(-gv)) * uv);
  }
}

// ---------------------------------------------------------------------------
extern "C" void kernel_launch(void* const* d_in, const int* in_sizes, int n_in,
                              void* d_out, int out_size, void* d_ws, size_t ws_size,
                              hipStream_t stream) {
  const float* x        = (const float*)d_in[0];
  const float* rope_cos = (const float*)d_in[1];
  const float* rope_sin = (const float*)d_in[2];
  const float* Wq       = (const float*)d_in[3];
  const float* Wk       = (const float*)d_in[4];
  const float* Wv       = (const float*)d_in[5];
  const float* Wo       = (const float*)d_in[6];
  const float* q_norm_w = (const float*)d_in[7];
  const float* k_norm_w = (const float*)d_in[8];
  const float* gate_W   = (const float*)d_in[9];
  const float* gate_b   = (const float*)d_in[10];
  const float* norm1_w  = (const float*)d_in[11];
  const float* norm2_w  = (const float*)d_in[12];
  const float* W_gate   = (const float*)d_in[13];
  const float* W_up     = (const float*)d_in[14];
  const float* W_down   = (const float*)d_in[15];
  const float* attn_sc  = (const float*)d_in[16];
  const float* mlp_sc   = (const float*)d_in[17];
  float* outp = (float*)d_out;

  char* base = (char*)d_ws;
  size_t off = 0;
  auto carve = [&](size_t bytes) -> char* {
    char* p = base + off;
    off += (bytes + 255) & ~(size_t)255;
    return p;
  };
  _Float16* h_half   = (_Float16*)carve((size_t)cNT * cH * 2);
  _Float16* q_half   = (_Float16*)carve((size_t)cNT * cH * 2);
  _Float16* k_half   = (_Float16*)carve((size_t)cNT * cH * 2);
  _Float16* v_half   = (_Float16*)carve((size_t)cNT * cH * 2);
  _Float16* ao_half  = (_Float16*)carve((size_t)cNT * cH * 2);
  _Float16* h2_half  = (_Float16*)carve((size_t)cNT * cH * 2);
  _Float16* g_half   = (_Float16*)carve((size_t)cNT * cMLP * 2);
  _Float16* u_half   = (_Float16*)carve((size_t)cNT * cMLP * 2);
  _Float16* mlp_half = (_Float16*)carve((size_t)cNT * cMLP * 2);
  float*    gate_f   = (float*)   carve((size_t)cNT * cNH * 4);
  float*    x1_f     = (float*)   carve((size_t)cNT * cH * 4);
  _Float16* Wq_h     = (_Float16*)carve((size_t)cH * cH * 2);
  _Float16* Wk_h     = (_Float16*)carve((size_t)cH * cH * 2);
  _Float16* Wv_h     = (_Float16*)carve((size_t)cH * cH * 2);
  _Float16* Wo_h     = (_Float16*)carve((size_t)cH * cH * 2);
  _Float16* Wg_h     = (_Float16*)carve((size_t)cH * cMLP * 2);
  _Float16* Wu_h     = (_Float16*)carve((size_t)cH * cMLP * 2);
  _Float16* Wd_h     = (_Float16*)carve((size_t)cMLP * cH * 2);
  if (off > ws_size) return;  // workspace too small; bail deterministically

  auto cvt = [&](const float* src, _Float16* dst, int n) {
    f32_to_f16_kernel<<<(n + 255) / 256, 256, 0, stream>>>(src, dst, n);
  };
  cvt(Wq, Wq_h, cH * cH);
  cvt(Wk, Wk_h, cH * cH);
  cvt(Wv, Wv_h, cH * cH);
  cvt(Wo, Wo_h, cH * cH);
  cvt(W_gate, Wg_h, cH * cMLP);
  cvt(W_up,   Wu_h, cH * cMLP);
  cvt(W_down, Wd_h, cMLP * cH);

  // 1) h = rmsnorm(x) * norm1_w
  rmsnorm_kernel<<<cNT, 256, 0, stream>>>(x, norm1_w, h_half);

  // 2) QKV projections
  dim3 gqkv(cH / 128, (cNT + 63) / 64);
  gemm_kernel<0><<<gqkv, 256, 0, stream>>>(h_half, Wq_h, q_half, nullptr, nullptr, nullptr, cNT, cH, cH);
  gemm_kernel<0><<<gqkv, 256, 0, stream>>>(h_half, Wk_h, k_half, nullptr, nullptr, nullptr, cNT, cH, cH);
  gemm_kernel<0><<<gqkv, 256, 0, stream>>>(h_half, Wv_h, v_half, nullptr, nullptr, nullptr, cNT, cH, cH);

  // 3) per-head rmsnorm + rope (+ 1/sqrt(HD) on q)
  int nwaves = cNT * cNH;
  qk_post_kernel<<<(nwaves + 3) / 4, 128, 0, stream>>>(q_half, q_norm_w, rope_cos, rope_sin, 0.125f);
  qk_post_kernel<<<(nwaves + 3) / 4, 128, 0, stream>>>(k_half, k_norm_w, rope_cos, rope_sin, 1.0f);

  // 4) sigmoid gate
  gate_kernel<<<(nwaves + 3) / 4, 128, 0, stream>>>(h_half, gate_W, gate_b, gate_f);

  // 5) flash attention + gating
  dim3 gattn((cS + 63) / 64, cNH, cB);
  attn_kernel<<<gattn, 128, 0, stream>>>(q_half, k_half, v_half, gate_f, ao_half);

  // 6) x1 = x + (attn @ Wo) * attn_scale
  gemm_kernel<1><<<gqkv, 256, 0, stream>>>(ao_half, Wo_h, nullptr, x1_f, x, attn_sc, cNT, cH, cH);

  // 7) h2 = rmsnorm(x1) * norm2_w
  rmsnorm_kernel<<<cNT, 256, 0, stream>>>(x1_f, norm2_w, h2_half);

  // 8) MLP gate/up projections
  dim3 gmlp(cMLP / 128, (cNT + 63) / 64);
  gemm_kernel<0><<<gmlp, 256, 0, stream>>>(h2_half, Wg_h, g_half, nullptr, nullptr, nullptr, cNT, cH, cMLP);
  gemm_kernel<0><<<gmlp, 256, 0, stream>>>(h2_half, Wu_h, u_half, nullptr, nullptr, nullptr, cNT, cH, cMLP);

  // 9) silu(g) * u
  int nmlp = cNT * cMLP;
  swiglu_kernel<<<(nmlp + 255) / 256, 256, 0, stream>>>(g_half, u_half, mlp_half, nmlp);

  // 10) out = x1 + (mlp @ W_down) * mlp_scale
  gemm_kernel<1><<<gqkv, 256, 0, stream>>>(mlp_half, Wd_h, nullptr, outp, x1_f, mlp_sc, cNT, cMLP, cH);
}
